// HSMSSD_20263655703007
// MI455X (gfx1250) — matrix operand order repaired and checked by reading, not compiled, not run on hardware
//
#include <hip/hip_runtime.h>

typedef __attribute__((ext_vector_type(16))) __bf16 v16bf;
typedef __attribute__((ext_vector_type(8)))  __bf16 v8bf;
typedef __attribute__((ext_vector_type(2)))  __bf16 v2bf;
typedef __attribute__((ext_vector_type(8)))  float  v8f;

// ---------- helpers ----------
__device__ __forceinline__ __bf16 f2bf(float f) { return (__bf16)f; }   // native v_cvt path

// pack two floats -> packed bf16 pair (v_cvt_pk_bf16_f32) and store as one b32
__device__ __forceinline__ void store_pk(__bf16* dst, float lo, float hi) {
    v2bf v; v[0] = f2bf(lo); v[1] = f2bf(hi);
    *(v2bf*)dst = v;
}

__device__ __forceinline__ v8f wmma_bf16(v16bf a, v16bf b, v8f c) {
    return __builtin_amdgcn_wmma_f32_16x16x32_bf16(
        /*neg_a=*/false, a, /*neg_b=*/false, b,
        /*c_mod=*/(short)0, c, /*reuse_a=*/false, /*reuse_b=*/false);
}

// A fragment from bf16 row-major source: 16 consecutive bf16 at p
__device__ __forceinline__ void load_a_bf(const __bf16* p, v16bf& d) {
    v8bf a0 = *(const v8bf*)p;
    v8bf a1 = *(const v8bf*)(p + 8);
#pragma unroll
    for (int e = 0; e < 8; ++e) { d[e] = a0[e]; d[8 + e] = a1[e]; }
}
// B fragment: 16 consecutive bf16 at p
__device__ __forceinline__ void load_b_bf(const __bf16* p, v16bf& d) {
    v8bf b0 = *(const v8bf*)p;
    v8bf b1 = *(const v8bf*)(p + 8);
#pragma unroll
    for (int e = 0; e < 8; ++e) { d[e] = b0[e]; d[8 + e] = b1[e]; }
}
// A fragment from fp32 row-major source: 16 consecutive floats at p
__device__ __forceinline__ void load_a_f32(const float* p, v16bf& d) {
    float4 v0 = *(const float4*)(p);
    float4 v1 = *(const float4*)(p + 4);
    float4 v2 = *(const float4*)(p + 8);
    float4 v3 = *(const float4*)(p + 12);
    d[0]=f2bf(v0.x); d[1]=f2bf(v0.y); d[2]=f2bf(v0.z); d[3]=f2bf(v0.w);
    d[4]=f2bf(v1.x); d[5]=f2bf(v1.y); d[6]=f2bf(v1.z); d[7]=f2bf(v1.w);
    d[8]=f2bf(v2.x); d[9]=f2bf(v2.y); d[10]=f2bf(v2.z); d[11]=f2bf(v2.w);
    d[12]=f2bf(v3.x); d[13]=f2bf(v3.y); d[14]=f2bf(v3.z); d[15]=f2bf(v3.w);
}
// B fragment from fp32: 16 consecutive floats at p
__device__ __forceinline__ void load_b_f32(const float* p, v16bf& d) {
    float4 v0 = *(const float4*)(p);
    float4 v1 = *(const float4*)(p + 4);
    float4 v2 = *(const float4*)(p + 8);
    float4 v3 = *(const float4*)(p + 12);
    d[0]=f2bf(v0.x); d[1]=f2bf(v0.y); d[2]=f2bf(v0.z); d[3]=f2bf(v0.w);
    d[4]=f2bf(v1.x); d[5]=f2bf(v1.y); d[6]=f2bf(v1.z); d[7]=f2bf(v1.w);
    d[8]=f2bf(v2.x); d[9]=f2bf(v2.y); d[10]=f2bf(v2.z); d[11]=f2bf(v2.w);
    d[12]=f2bf(v3.x); d[13]=f2bf(v3.y); d[14]=f2bf(v3.z); d[15]=f2bf(v3.w);
}

// ---------- small prep kernels ----------
__global__ void k_f2bf(const float* __restrict__ s, __bf16* __restrict__ d, int n) {
    int i = blockIdx.x * 256 + threadIdx.x;
    if (i < n) d[i] = f2bf(s[i]);
}

// src [B, C, S] fp32 -> dst [B, S, C] bf16  (C*S == gridDim.x*256)
__global__ void k_transpose_bf(const float* __restrict__ src, __bf16* __restrict__ dst,
                               int C, int S) {
    int b = blockIdx.y;
    int i = blockIdx.x * 256 + threadIdx.x;
    if (i < C * S) {
        int c = i / S, s = i % S;
        dst[((size_t)b * S + s) * C + c] = f2bf(src[((size_t)b * C + c) * S + s]);
    }
}

// ---------- G1: BCdt = W_bcdt @ x + b   [16,192,4096] ----------
// double-buffered LDS tile; pair-packed (cvt_pk + ds_store_b32) transpose staging
__global__ __launch_bounds__(256) void k_gemm_bcdt(
    const float* __restrict__ x,    // [B,512,4096]
    const __bf16* __restrict__ Wb,  // [192,512] bf16
    const float* __restrict__ bias, // [192]
    float* __restrict__ out)        // [B,192,4096]
{
    const int b = blockIdx.z, l0 = blockIdx.x * 64, m0 = blockIdx.y * 32;
    const int t = threadIdx.x, w = t >> 5, lane = t & 31;
    const int g = lane >> 4, ln = lane & 15;
    const int mi = w & 1, ni = w >> 1;
    const int mBase = m0 + 16 * mi, nBase = 16 * ni;

    __shared__ __bf16 sB[2][64][72];   // [buf][n][k], K=64 chunk, padded stride

    const float* xb = x + (size_t)b * 512 * 4096 + l0;
    const int rp2 = 2 * (t >> 3);      // even K row within chunk: 0..62
    const int n0 = (t & 7) * 8;        // 0..56

    float4 r0, r1, r2, r3;             // rows rp2 (r0,r1) and rp2+1 (r2,r3)
    {
        const float* p = xb + (size_t)rp2 * 4096 + n0;
        r0 = *(const float4*)(p);        r1 = *(const float4*)(p + 4);
        r2 = *(const float4*)(p + 4096); r3 = *(const float4*)(p + 4100);
    }
    auto stage = [&](int buf) {
        store_pk(&sB[buf][n0+0][rp2], r0.x, r2.x);
        store_pk(&sB[buf][n0+1][rp2], r0.y, r2.y);
        store_pk(&sB[buf][n0+2][rp2], r0.z, r2.z);
        store_pk(&sB[buf][n0+3][rp2], r0.w, r2.w);
        store_pk(&sB[buf][n0+4][rp2], r1.x, r3.x);
        store_pk(&sB[buf][n0+5][rp2], r1.y, r3.y);
        store_pk(&sB[buf][n0+6][rp2], r1.z, r3.z);
        store_pk(&sB[buf][n0+7][rp2], r1.w, r3.w);
    };
    stage(0);
    int cur = 0;
    v8f acc = {};

    for (int kc = 0; kc < 512; kc += 64) {
        const bool has_next = (kc + 64) < 512;
        if (has_next) {    // issue next chunk's global loads before the barrier
            const float* p = xb + (size_t)(kc + 64 + rp2) * 4096 + n0;
            r0 = *(const float4*)(p);        r1 = *(const float4*)(p + 4);
            r2 = *(const float4*)(p + 4096); r3 = *(const float4*)(p + 4100);
        }
        __syncthreads();
#pragma unroll
        for (int kcc = 0; kcc < 64; kcc += 32) {
            v16bf afrag, bfrag;
            load_a_bf(Wb + (size_t)(mBase + ln) * 512 + kc + kcc + 16 * g, afrag);
            load_b_bf(&sB[cur][nBase + ln][kcc + 16 * g], bfrag);
            acc = wmma_bf16(afrag, bfrag, acc);
        }
        if (has_next) stage(cur ^ 1);
        cur ^= 1;
    }
    float* ob = out + (size_t)b * 192 * 4096 + l0 + nBase + ln;
#pragma unroll
    for (int j = 0; j < 8; ++j) {
        int m = mBase + j + 8 * g;
        ob[(size_t)m * 4096] = acc[j] + bias[m];
    }
}

// ---------- depthwise 3x3 SAME conv on [B,192,64,64] ----------
__global__ void k_dwconv(const float* __restrict__ in, const float* __restrict__ Wd,
                         const float* __restrict__ bd, float* __restrict__ out)
{
    int idx = blockIdx.x * 256 + threadIdx.x;
    if (idx >= 16 * 192 * 4096) return;
    int wc = idx & 63;
    int hr = (idx >> 6) & 63;
    int bc = idx >> 12;            // b*192 + ch
    int ch = bc % 192;
    const float* ip = in + ((size_t)bc << 12);
    const float* wp = Wd + ch * 9;
    float acc = bd[ch];
#pragma unroll
    for (int dy = -1; dy <= 1; ++dy) {
        int y = hr + dy;
        if ((unsigned)y >= 64u) continue;
#pragma unroll
        for (int dx = -1; dx <= 1; ++dx) {
            int xx = wc + dx;
            if ((unsigned)xx >= 64u) continue;
            acc += ip[(y << 6) + xx] * wp[(dy + 1) * 3 + (dx + 1)];
        }
    }
    out[idx] = acc;
}

// ---------- softmax over L (with +A[s]) * Bm -> AB  [B,64,4096] ----------
__global__ __launch_bounds__(256) void k_softmax_ab(
    const float* __restrict__ bcdt,  // [B,192,4096] (post conv)
    const float* __restrict__ A,     // [64]
    float* __restrict__ AB)          // [B,64,4096]
{
    int s = blockIdx.x, b = blockIdx.y, t = threadIdx.x;
    const float* Bm = bcdt + ((size_t)b * 192 + s) * 4096;
    const float* dt = bcdt + ((size_t)b * 192 + 128 + s) * 4096;
    float* ab = AB + ((size_t)b * 64 + s) * 4096;
    float Ab = A[s];
    __shared__ float red[8];

    float v[16];
    float mx = -3.4e38f;
#pragma unroll
    for (int r = 0; r < 16; ++r) {
        v[r] = dt[t + 256 * r] + Ab;
        mx = fmaxf(mx, v[r]);
    }
    for (int off = 16; off; off >>= 1) mx = fmaxf(mx, __shfl_xor(mx, off, 32));
    if ((t & 31) == 0) red[t >> 5] = mx;
    __syncthreads();
    if (t == 0) { float m = red[0]; for (int i = 1; i < 8; ++i) m = fmaxf(m, red[i]); red[0] = m; }
    __syncthreads();
    mx = red[0];
    __syncthreads();

    float sum = 0.f;
#pragma unroll
    for (int r = 0; r < 16; ++r) { v[r] = __expf(v[r] - mx); sum += v[r]; }
    for (int off = 16; off; off >>= 1) sum += __shfl_xor(sum, off, 32);
    if ((t & 31) == 0) red[t >> 5] = sum;
    __syncthreads();
    if (t == 0) { float sm = 0.f; for (int i = 0; i < 8; ++i) sm += red[i]; red[0] = sm; }
    __syncthreads();
    float inv = 1.0f / red[0];
#pragma unroll
    for (int r = 0; r < 16; ++r)
        ab[t + 256 * r] = v[r] * inv * Bm[t + 256 * r];
}

// ---------- G2: h[b,c,s] = sum_l x[b,c,l]*AB[b,s,l]   [B,512,64] ----------
// block: M=64 (4 waves), each wave owns one M sub-tile and two N tiles (A reuse)
__global__ __launch_bounds__(256) void k_gemm_h(
    const float* __restrict__ x, const float* __restrict__ AB,
    float* __restrict__ h)
{
    const int b = blockIdx.y, m0 = blockIdx.x * 64;
    const int t = threadIdx.x, w = t >> 5, lane = t & 31;
    const int g = lane >> 4, ln = lane & 15;
    const int mi = w & 3, nh = w >> 2;          // M sub-tile, N half
    const int mBase = m0 + 16 * mi;

    const float* ar  = x  + ((size_t)b * 512 + mBase + ln) * 4096;          // A row (M=c, K=l)
    const float* br0 = AB + ((size_t)b * 64 + 32 * nh + ln) * 4096;         // B^T rows
    const float* br1 = AB + ((size_t)b * 64 + 32 * nh + 16 + ln) * 4096;
    v8f acc0 = {}, acc1 = {};
#pragma unroll 2
    for (int kc = 0; kc < 4096; kc += 32) {
        v16bf af, bf0, bf1;
        load_a_f32(ar + kc + 16 * g, af);
        load_b_f32(br0 + kc + 16 * g, bf0);
        load_b_f32(br1 + kc + 16 * g, bf1);
        acc0 = wmma_bf16(af, bf0, acc0);
        acc1 = wmma_bf16(af, bf1, acc1);
        if (kc + 64 < 4096)            // keep the streaming A rows ahead (global_prefetch_b8)
            __builtin_prefetch(ar + kc + 64 + 8 * g, 0, 3);
    }
    float* hp = h + (size_t)b * 512 * 64;
#pragma unroll
    for (int j = 0; j < 8; ++j) {
        size_t row = (size_t)(mBase + j + 8 * g) * 64;
        hp[row + 32 * nh + ln]      = acc0[j];
        hp[row + 32 * nh + 16 + ln] = acc1[j];
    }
}

// ---------- G3: hz = W_hz @ h + b_hz   [B,1024,64] ----------
__global__ __launch_bounds__(256) void k_gemm_hz(
    const __bf16* __restrict__ Whz,   // [1024,512]
    const __bf16* __restrict__ hT,    // [B,64,512]
    const float* __restrict__ bhz, float* __restrict__ hz)
{
    const int b = blockIdx.y, m0 = blockIdx.x * 32;
    const int t = threadIdx.x, w = t >> 5, lane = t & 31;
    const int g = lane >> 4, ln = lane & 15;
    const int mi = w & 1, ni = w >> 1;
    const int mBase = m0 + 16 * mi, nBase = 16 * ni;

    const __bf16* ar = Whz + (size_t)(mBase + ln) * 512;
    const __bf16* br = hT + ((size_t)b * 64 + nBase + ln) * 512;
    v8f acc = {};
#pragma unroll 2
    for (int kc = 0; kc < 512; kc += 32) {
        v16bf afrag, bfrag;
        load_a_bf(ar + kc + 16 * g, afrag);
        load_b_bf(br + kc + 16 * g, bfrag);
        acc = wmma_bf16(afrag, bfrag, acc);
    }
    float* op = hz + (size_t)b * 1024 * 64 + nBase + ln;
#pragma unroll
    for (int j = 0; j < 8; ++j) {
        int m = mBase + j + 8 * g;
        op[(size_t)m * 64] = acc[j] + bhz[m];
    }
}

// ---------- gate: hg = h1*(silu(z)+D) -> transposed bf16 [B,64,512] ----------
__global__ void k_gate(const float* __restrict__ hz, const float* __restrict__ Dp,
                       __bf16* __restrict__ hgT)
{
    int i = blockIdx.x * 256 + threadIdx.x;   // over 16*512*64
    if (i >= 16 * 512 * 64) return;
    int s = i & 63, c = (i >> 6) & 511, b = i >> 15;
    const float* base = hz + (size_t)b * 1024 * 64;
    float h1 = base[(size_t)c * 64 + s];
    float z  = base[(size_t)(512 + c) * 64 + s];
    float sil = z / (1.f + __expf(-z));
    hgT[((size_t)b * 64 + s) * 512 + c] = f2bf(h1 * (sil + Dp[0]));
}

// ---------- G4+G5: h_final = W_out@hg + b_out + W_prev@h_prev + b_prev ----------
__global__ __launch_bounds__(256) void k_gemm_out(
    const __bf16* __restrict__ Wout, const __bf16* __restrict__ Wprev,
    const __bf16* __restrict__ hgT,  const __bf16* __restrict__ hprevT,
    const float* __restrict__ bout,  const float* __restrict__ bprev,
    float* __restrict__ h_out, __bf16* __restrict__ hfinB)
{
    const int b = blockIdx.y, m0 = blockIdx.x * 32;
    const int t = threadIdx.x, w = t >> 5, lane = t & 31;
    const int g = lane >> 4, ln = lane & 15;
    const int mi = w & 1, ni = w >> 1;
    const int mBase = m0 + 16 * mi, nBase = 16 * ni;

    const __bf16* a1 = Wout  + (size_t)(mBase + ln) * 512;
    const __bf16* a2 = Wprev + (size_t)(mBase + ln) * 512;
    const __bf16* b1 = hgT    + ((size_t)b * 64 + nBase + ln) * 512;
    const __bf16* b2 = hprevT + ((size_t)b * 64 + nBase + ln) * 512;
    v8f acc = {};
    for (int kc = 0; kc < 512; kc += 32) {
        v16bf af, bf;
        load_a_bf(a1 + kc + 16 * g, af);
        load_b_bf(b1 + kc + 16 * g, bf);
        acc = wmma_bf16(af, bf, acc);
        load_a_bf(a2 + kc + 16 * g, af);
        load_b_bf(b2 + kc + 16 * g, bf);
        acc = wmma_bf16(af, bf, acc);
    }
#pragma unroll
    for (int j = 0; j < 8; ++j) {
        int m = mBase + j + 8 * g;
        float v = acc[j] + bout[m] + bprev[m];
        size_t o = ((size_t)b * 512 + m) * 64 + nBase + ln;
        h_out[o] = v;
        hfinB[o] = f2bf(v);
    }
}

// ---------- G6: y[b,c,l] = sum_s h[b,c,s]*Cm[b,s,l]   [B,512,4096] ----------
__global__ __launch_bounds__(256) void k_gemm_y(
    const __bf16* __restrict__ hfin,     // [B,512,64]
    const float* __restrict__ bcdt,      // [B,192,4096]; Cm at channels 64..127
    float* __restrict__ y)
{
    const int b = blockIdx.z, l0 = blockIdx.x * 64, m0 = blockIdx.y * 32;
    const int t = threadIdx.x, w = t >> 5, lane = t & 31;
    const int g = lane >> 4, ln = lane & 15;
    const int mi = w & 1, ni = w >> 1;
    const int mBase = m0 + 16 * mi, nBase = 16 * ni;

    __shared__ __bf16 sB[64][72];   // [n][k], K=64, padded stride

    const float* cm = bcdt + ((size_t)b * 192 + 64) * 4096 + l0;
    {   // stage Cm tile transposed: pair-packed (cvt_pk + ds_store_b32)
        const int rp2 = 2 * (t >> 3);     // even K row: 0..62
        const int n0 = (t & 7) * 8;       // 0..56
        const float* p = cm + (size_t)rp2 * 4096 + n0;
        float4 r0 = *(const float4*)(p);        float4 r1 = *(const float4*)(p + 4);
        float4 r2 = *(const float4*)(p + 4096); float4 r3 = *(const float4*)(p + 4100);
        store_pk(&sB[n0+0][rp2], r0.x, r2.x);
        store_pk(&sB[n0+1][rp2], r0.y, r2.y);
        store_pk(&sB[n0+2][rp2], r0.z, r2.z);
        store_pk(&sB[n0+3][rp2], r0.w, r2.w);
        store_pk(&sB[n0+4][rp2], r1.x, r3.x);
        store_pk(&sB[n0+5][rp2], r1.y, r3.y);
        store_pk(&sB[n0+6][rp2], r1.z, r3.z);
        store_pk(&sB[n0+7][rp2], r1.w, r3.w);
    }
    __syncthreads();
    const __bf16* ar = hfin + ((size_t)b * 512 + mBase + ln) * 64;
    v8f acc = {};
#pragma unroll
    for (int kc = 0; kc < 64; kc += 32) {
        v16bf afrag, bfrag;
        load_a_bf(ar + kc + 16 * g, afrag);
        load_b_bf(&sB[nBase + ln][kc + 16 * g], bfrag);
        acc = wmma_bf16(afrag, bfrag, acc);
    }
    float* yp = y + (size_t)b * 512 * 4096 + l0 + nBase + ln;
#pragma unroll
    for (int j = 0; j < 8; ++j)
        yp[(size_t)(mBase + j + 8 * g) * 4096] = acc[j];
}

// ---------- launch ----------
extern "C" void kernel_launch(void* const* d_in, const int* in_sizes, int n_in,
                              void* d_out, int out_size, void* d_ws, size_t ws_size,
                              hipStream_t stream) {
    (void)in_sizes; (void)n_in; (void)out_size; (void)ws_size;
    const float* x      = (const float*)d_in[0];
    const float* h_prev = (const float*)d_in[1];
    const float* W_bcdt = (const float*)d_in[2];
    const float* b_bcdt = (const float*)d_in[3];
    const float* W_dw   = (const float*)d_in[4];
    const float* b_dw   = (const float*)d_in[5];
    const float* W_hz   = (const float*)d_in[6];
    const float* b_hz   = (const float*)d_in[7];
    const float* W_out  = (const float*)d_in[8];
    const float* b_out  = (const float*)d_in[9];
    const float* W_prev = (const float*)d_in[10];
    const float* b_prev = (const float*)d_in[11];
    const float* A      = (const float*)d_in[12];
    const float* Dsc    = (const float*)d_in[13];

    float* y_out = (float*)d_out;                               // [16,512,4096]
    float* h_out = y_out + (size_t)16 * 512 * 4096;             // [16,512,64]

    char* ws = (char*)d_ws;
    constexpr size_t S_pre = (size_t)16 * 192 * 4096 * 4;
    float*  bcdt_pre  = (float*)(ws);                // 50.3 MB
    float*  bcdt_post = (float*)(ws + S_pre);        // 50.3 MB
    float*  ABw       = (float*)(ws);                // reuses pre region after dwconv
    size_t off = 2 * S_pre;
    float*  h_ws   = (float*)(ws + off);  off += (size_t)16 * 512 * 64 * 4;
    __bf16* hT     = (__bf16*)(ws + off); off += (size_t)16 * 64 * 512 * 2;
    float*  hz_ws  = (float*)(ws + off);  off += (size_t)16 * 1024 * 64 * 4;
    __bf16* hgT    = (__bf16*)(ws + off); off += (size_t)16 * 64 * 512 * 2;
    __bf16* hprevT = (__bf16*)(ws + off); off += (size_t)16 * 64 * 512 * 2;
    __bf16* hfinB  = (__bf16*)(ws + off); off += (size_t)16 * 512 * 64 * 2;
    __bf16* WbB    = (__bf16*)(ws + off); off += (size_t)192 * 512 * 2;
    __bf16* WhzB   = (__bf16*)(ws + off); off += (size_t)1024 * 512 * 2;
    __bf16* WoutB  = (__bf16*)(ws + off); off += (size_t)512 * 512 * 2;
    __bf16* WprevB = (__bf16*)(ws + off); off += (size_t)512 * 512 * 2;

    // weight conversions (fp32 -> bf16)
    k_f2bf<<<(192 * 512 + 255) / 256, 256, 0, stream>>>(W_bcdt, WbB, 192 * 512);
    k_f2bf<<<(1024 * 512 + 255) / 256, 256, 0, stream>>>(W_hz, WhzB, 1024 * 512);
    k_f2bf<<<(512 * 512 + 255) / 256, 256, 0, stream>>>(W_out, WoutB, 512 * 512);
    k_f2bf<<<(512 * 512 + 255) / 256, 256, 0, stream>>>(W_prev, WprevB, 512 * 512);
    k_transpose_bf<<<dim3(128, 16), 256, 0, stream>>>(h_prev, hprevT, 512, 64);

    // G1: BCdt projection
    k_gemm_bcdt<<<dim3(64, 6, 16), 256, 0, stream>>>(x, WbB, b_bcdt, bcdt_pre);
    // depthwise conv
    k_dwconv<<<(16 * 192 * 4096) / 256, 256, 0, stream>>>(bcdt_pre, W_dw, b_dw, bcdt_post);
    // softmax(dt + A) * Bm -> AB  (overwrites pre region)
    k_softmax_ab<<<dim3(64, 16), 256, 0, stream>>>(bcdt_post, A, ABw);
    // G2: h = x @ AB^T
    k_gemm_h<<<dim3(8, 16), 256, 0, stream>>>(x, ABw, h_ws);
    k_transpose_bf<<<dim3(128, 16), 256, 0, stream>>>(h_ws, hT, 512, 64);
    // G3: hz projection
    k_gemm_hz<<<dim3(32, 16), 256, 0, stream>>>(WhzB, hT, b_hz, hz_ws);
    // gate
    k_gate<<<(16 * 512 * 64) / 256, 256, 0, stream>>>(hz_ws, Dsc, hgT);
    // G4+G5: output projection + prev-state projection
    k_gemm_out<<<dim3(16, 16), 256, 0, stream>>>(WoutB, WprevB, hgT, hprevT,
                                                 b_out, b_prev, h_out, hfinB);
    // G6: y = h @ Cm
    k_gemm_y<<<dim3(64, 16, 16), 256, 0, stream>>>(hfinB, bcdt_post, y_out);
}
